// SNNIDEModule_86706799772117
// MI455X (gfx1250) — compile-verified, matrix operand order checked by reading, not run on hardware
//
#include <hip/hip_runtime.h>

// ---------------------------------------------------------------------------
// SNN implicit-equilibrium forward for MI455X (gfx1250, wave32, WMMA bf16).
//   x   = u @ Fm.T + b                       (bf16 WMMA GEMM, f32 accum)
//   loop T=30: v += x + s@W.T ; spike ; reset (persistent per-32-row block)
//   out = acc / T
// Spikes are exactly {0,1} -> lossless in bf16; W/Fm/u rounded once to bf16.
// W (2MB bf16) is L2-resident; each B fragment is reused for 2 row-tiles
// (M=32 per WG) -> 32 flop/byte against L2. All 4 B fragments of a K-chunk
// are loaded into distinct registers before the WMMA burst so their load
// clauses overlap (staggered loadcnt waits instead of wait-0 per tile).
// ---------------------------------------------------------------------------

typedef __attribute__((ext_vector_type(16))) __bf16 v16bf;
typedef __attribute__((ext_vector_type(8)))  __bf16 v8bf;
typedef __attribute__((ext_vector_type(8)))  float  v8f;

#define BATCH   4096
#define DIN     4096
#define DH      1024
#define TSTEPS  30
#define VTH     1.0f

#define SSTR    1032          // LDS spike row stride (bf16 elems), bank-padded

// A/B fragment loader for v_wmma_f32_16x16x32_bf16.
// Per-lane 16 bf16 values: elements 0-7 at base+hi8, elements 8-15 at
// base+16+hi8  (hi8 = 8 for lanes 16-31, 0 for lanes 0-15).
__device__ __forceinline__ v16bf load_frag(const __bf16* base, int hi8) {
    union { v16bf v; v8bf h[2]; } u;
    u.h[0] = *reinterpret_cast<const v8bf*>(base + hi8);
    u.h[1] = *reinterpret_cast<const v8bf*>(base + 16 + hi8);
    return u.v;
}

// ---------------------------------------------------------------------------
// Kernel 0: f32 -> bf16 conversion (W and Fm stay row-major; B fragments are
// read N-major, which is exactly a row of W / Fm -> no transpose needed).
// ---------------------------------------------------------------------------
__global__ void cvt_f32_to_bf16(const float* __restrict__ src,
                                __bf16* __restrict__ dst, int n) {
    int i = blockIdx.x * blockDim.x + threadIdx.x;
    if (i < n) dst[i] = (__bf16)src[i];
}

// ---------------------------------------------------------------------------
// Kernel 1: x = u @ Fm.T + b.   Grid: 128 WGs x 512 thr (16 waves).
// WG owns 32 rows of x; wave w owns 64 columns (4 WMMA tiles) x 2 row-tiles.
// A (u) staged f32->bf16 through LDS (32x64 per round); B streams from bf16
// Fm and is reused for both row-tiles.
// ---------------------------------------------------------------------------
__global__ __launch_bounds__(512)
void gemm_x_kernel(const float* __restrict__ u, const __bf16* __restrict__ Fmb,
                   const float* __restrict__ b, float* __restrict__ x) {
    __shared__ __align__(16) __bf16 sA[32 * 72];   // 32 rows x 64 K, stride 72

    const int tid  = threadIdx.x;
    const int lane = tid & 31;
    const int wv   = tid >> 5;          // 0..15
    const int lo   = lane & 15;
    const int hi8  = (lane >> 4) << 3;  // 0 or 8
    const int r0   = blockIdx.x * 32;

    v8f c[2][4] = {};

    for (int k0 = 0; k0 < DIN; k0 += 64) {
        // stage 32x64 tile of u as bf16 (4 elements per thread, coalesced)
#pragma unroll
        for (int i = 0; i < 4; ++i) {
            int e   = tid + i * 512;
            int row = e >> 6, col = e & 63;
            sA[row * 72 + col] =
                (__bf16)u[(size_t)(r0 + row) * DIN + k0 + col];
        }
        __syncthreads();

#pragma unroll
        for (int kk = 0; kk < 64; kk += 32) {
            // batch-load all B fragments first: 4 load clauses in flight
            v16bf bb[4];
#pragma unroll
            for (int t = 0; t < 4; ++t) {
                int n = wv * 64 + t * 16 + lo;
                bb[t] = load_frag(&Fmb[(size_t)n * DIN + k0 + kk], hi8);
            }
            v16bf a0 = load_frag(&sA[lo * 72 + kk], hi8);
            v16bf a1 = load_frag(&sA[(16 + lo) * 72 + kk], hi8);
#pragma unroll
            for (int t = 0; t < 4; ++t) {
                c[0][t] = __builtin_amdgcn_wmma_f32_16x16x32_bf16(
                    false, a0, false, bb[t], (short)0, c[0][t], false, false);
                c[1][t] = __builtin_amdgcn_wmma_f32_16x16x32_bf16(
                    false, a1, false, bb[t], (short)0, c[1][t], false, false);
            }
        }
        __syncthreads();
    }

#pragma unroll
    for (int mt = 0; mt < 2; ++mt)
#pragma unroll
        for (int t = 0; t < 4; ++t) {
            int n = wv * 64 + t * 16 + lo;
            float bias = b[n];
#pragma unroll
            for (int j = 0; j < 8; ++j) {
                int m = mt * 16 + j + hi8;   // C layout: vgpr j -> M=j (+8 hi)
                x[(size_t)(r0 + m) * DH + n] = c[mt][t][j] + bias;
            }
        }
}

// ---------------------------------------------------------------------------
// Kernel 2: persistent LIF loop.  Grid: 128 WGs x 512 thr (16 waves).
// WG owns 32 batch rows for all 30 steps. Spikes (bf16) live in LDS,
// v / acc live in VGPRs, W streams from L2 (each B frag reused 2x),
// x re-read from L2 each step.
// ---------------------------------------------------------------------------
__global__ __launch_bounds__(512)
void snn_loop_kernel(const float* __restrict__ x, const __bf16* __restrict__ Wb,
                     float* __restrict__ out) {
    __shared__ __align__(16) __bf16 sS[32 * SSTR];  // 32 x 1024 spikes, padded

    const int tid  = threadIdx.x;
    const int lane = tid & 31;
    const int wv   = tid >> 5;
    const int lo   = lane & 15;
    const int hi8  = (lane >> 4) << 3;
    const int r0   = blockIdx.x * 32;

    // zero spike buffer (s = 0 at t = 0)
    unsigned int* sz = reinterpret_cast<unsigned int*>(sS);
    for (int i = tid; i < (32 * SSTR) / 2; i += 512) sz[i] = 0u;

    float v[2][4][8], acc[2][4][8];
#pragma unroll
    for (int mt = 0; mt < 2; ++mt)
#pragma unroll
        for (int t = 0; t < 4; ++t)
#pragma unroll
            for (int j = 0; j < 8; ++j) {
                v[mt][t][j]   = 0.0f;
                acc[mt][t][j] = 0.0f;
            }
    __syncthreads();

    for (int step = 0; step < TSTEPS; ++step) {
        // fb = s @ W.T for this wave's 32x64 output slab
        v8f c[2][4] = {};
        for (int k0 = 0; k0 < DH; k0 += 32) {
            // batch-load all B fragments first: 4 load clauses in flight
            v16bf bb[4];
#pragma unroll
            for (int t = 0; t < 4; ++t) {
                int n = wv * 64 + t * 16 + lo;
                bb[t] = load_frag(&Wb[(size_t)n * DH + k0], hi8);
            }
            v16bf a0 = load_frag(&sS[lo * SSTR + k0], hi8);
            v16bf a1 = load_frag(&sS[(16 + lo) * SSTR + k0], hi8);
#pragma unroll
            for (int t = 0; t < 4; ++t) {
                c[0][t] = __builtin_amdgcn_wmma_f32_16x16x32_bf16(
                    false, a0, false, bb[t], (short)0, c[0][t], false, false);
                c[1][t] = __builtin_amdgcn_wmma_f32_16x16x32_bf16(
                    false, a1, false, bb[t], (short)0, c[1][t], false, false);
            }
        }
        __syncthreads();   // everyone done reading sS

        // LIF update + publish new spikes (x streamed from L2)
#pragma unroll
        for (int mt = 0; mt < 2; ++mt)
#pragma unroll
            for (int t = 0; t < 4; ++t) {
                int n = wv * 64 + t * 16 + lo;
#pragma unroll
                for (int j = 0; j < 8; ++j) {
                    int m = mt * 16 + j + hi8;
                    float vn = v[mt][t][j] + x[(size_t)(r0 + m) * DH + n]
                             + c[mt][t][j];
                    float sp = (vn >= VTH) ? 1.0f : 0.0f;
                    v[mt][t][j]   = vn - VTH * sp;
                    acc[mt][t][j] += sp;
                    sS[m * SSTR + n] = (__bf16)sp;
                }
            }
        __syncthreads();   // spikes visible before next step's reads
    }

    const float inv_t = 1.0f / (float)TSTEPS;
#pragma unroll
    for (int mt = 0; mt < 2; ++mt)
#pragma unroll
        for (int t = 0; t < 4; ++t) {
            int n = wv * 64 + t * 16 + lo;
#pragma unroll
            for (int j = 0; j < 8; ++j) {
                int m = mt * 16 + j + hi8;
                out[(size_t)(r0 + m) * DH + n] = acc[mt][t][j] * inv_t;
            }
        }
}

// ---------------------------------------------------------------------------
extern "C" void kernel_launch(void* const* d_in, const int* in_sizes, int n_in,
                              void* d_out, int out_size, void* d_ws, size_t ws_size,
                              hipStream_t stream) {
    const float* u  = (const float*)d_in[0];   // [4096, 4096]
    const float* W  = (const float*)d_in[1];   // [1024, 1024]
    const float* Fm = (const float*)d_in[2];   // [1024, 4096]
    const float* b  = (const float*)d_in[3];   // [1024]
    float* out = (float*)d_out;                // [4096, 1024] f32

    // workspace layout: x (16MB f32) | Fm bf16 (8MB) | W bf16 (2MB)
    char* ws = (char*)d_ws;
    const size_t xBytes  = (size_t)BATCH * DH * sizeof(float);
    const size_t fmBytes = (size_t)DH * DIN * sizeof(__bf16);
    float*  x   = (float*)ws;
    __bf16* Fmb = (__bf16*)(ws + xBytes);
    __bf16* Wb  = (__bf16*)(ws + xBytes + fmBytes);

    {
        int n = DH * DIN;
        cvt_f32_to_bf16<<<(n + 255) / 256, 256, 0, stream>>>(Fm, Fmb, n);
    }
    {
        int n = DH * DH;
        cvt_f32_to_bf16<<<(n + 255) / 256, 256, 0, stream>>>(W, Wb, n);
    }

    gemm_x_kernel<<<BATCH / 32, 512, 0, stream>>>(u, Fmb, b, x);
    snn_loop_kernel<<<BATCH / 32, 512, 0, stream>>>(x, Wb, out);
}
